// RQKmeansTokenizer_3229815407337
// MI455X (gfx1250) — matrix operand order repaired and checked by reading, not compiled, not run on hardware
//
#include <hip/hip_runtime.h>

// ---- vector types ----
typedef __attribute__((ext_vector_type(16))) __bf16 v16bf;
typedef __attribute__((ext_vector_type(8)))  __bf16 v8bf;
typedef __attribute__((ext_vector_type(4)))  __bf16 v4bf;
typedef __attribute__((ext_vector_type(8)))  float  v8f;
typedef __attribute__((ext_vector_type(4)))  unsigned int v4u;
typedef __attribute__((ext_vector_type(8)))  int    v8i;
typedef __attribute__((ext_vector_type(4)))  int    v4i;

#define BB 128
#define NN 1024
#define DD 512
#define LL 3
#define KK 256

#define TOK_PER_WG 128   // 8 waves x 16 tokens
#define THREADS    256

// ---------------------------------------------------------------------------
// Prep kernel: codebook f32 -> bf16 into d_ws, plus ||c||^2 per centroid.
// One wave per centroid row (grid = L*K).
// ---------------------------------------------------------------------------
__global__ __launch_bounds__(32) void rq_prep(const float* __restrict__ cb,
                                              __bf16* __restrict__ cbbf,
                                              float* __restrict__ csqw)
{
    const int row  = blockIdx.x;      // 0 .. L*K-1
    const int lane = threadIdx.x;     // 0 .. 31
    const float4* src = (const float4*)(cb + (size_t)row * DD);
    __bf16* dst = cbbf + (size_t)row * DD;
    float ss = 0.f;
    #pragma unroll
    for (int j = 0; j < 4; ++j) {
        int d4 = lane + 32 * j;
        float4 f = src[d4];
        ss += f.x * f.x + f.y * f.y + f.z * f.z + f.w * f.w;
        v4bf o;
        o[0] = (__bf16)f.x; o[1] = (__bf16)f.y;
        o[2] = (__bf16)f.z; o[3] = (__bf16)f.w;
        *(v4bf*)(dst + d4 * 4) = o;
    }
    #pragma unroll
    for (int m = 1; m < 32; m <<= 1) ss += __shfl_xor(ss, m, 32);
    if (lane == 0) csqw[row] = ss;
}

// ---------------------------------------------------------------------------
// Tensor Data Mover: DMA one codebook d-chunk (256 rows x 64B, stride 1024B)
// from the pre-converted bf16 codebook into LDS.  D# per CDNA5 ISA ch.8.
// ---------------------------------------------------------------------------
__device__ __forceinline__ void tdm_issue_chunk(const __bf16* cbbf, int layer,
                                                int chunk, unsigned ldsOff)
{
    unsigned long long gaddr =
        (unsigned long long)(uintptr_t)(cbbf + ((size_t)layer * KK * DD + (size_t)chunk * 32));
    v4u g0;
    g0[0] = 1u;                                   // count=1, user mode
    g0[1] = ldsOff;                               // LDS byte address
    g0[2] = (unsigned)(gaddr & 0xffffffffu);
    g0[3] = (unsigned)((gaddr >> 32) & 0x1ffffffu) | (2u << 30);  // type=2
    // group1 packed as 4 qwords (bit positions per ISA table):
    // data_size=3 (8B units), tensor_dim0=8, tensor_dim1=256,
    // tile_dim0=8, tile_dim1=256, tensor_dim0_stride=128
    unsigned long long q0 = (3ull << 16) | (8ull << 48);
    unsigned long long q1 = (256ull << 16) | (8ull << 48);
    unsigned long long q2 = 256ull | (128ull << 32);
    unsigned long long q3 = 0ull;
    v8i g1;
    g1[0] = (int)q0; g1[1] = (int)(q0 >> 32);
    g1[2] = (int)q1; g1[3] = (int)(q1 >> 32);
    g1[4] = (int)q2; g1[5] = (int)(q2 >> 32);
    g1[6] = (int)q3; g1[7] = (int)(q3 >> 32);
    v4i gz = {0, 0, 0, 0};
#if __clang_major__ >= 23
    v8i gz8 = {0, 0, 0, 0, 0, 0, 0, 0};
    __builtin_amdgcn_tensor_load_to_lds(g0, g1, gz, gz, gz8, 0);
#else
    __builtin_amdgcn_tensor_load_to_lds(g0, g1, gz, gz, 0);
#endif
}

// ---------------------------------------------------------------------------
// Fused residual-quantization kernel.
// PRECONV=true : bf16 codebook + csq live in d_ws, chunks staged by TDM,
//                double-buffered, DMA overlapped with WMMA.
// PRECONV=false: self-converting staging (workspace too small fallback).
// Occupancy is LDS-limited (1 WG/WGP -> 2 waves/SIMD), so relax the VGPR
// budget (launch_bounds min-waves=1) to let multiple B fragments stay live.
// ---------------------------------------------------------------------------
template <bool PRECONV>
__global__ __launch_bounds__(THREADS, 1) void rq_fused(const float* __restrict__ feat,
                                                       const float* __restrict__ cb,
                                                       const __bf16* __restrict__ cbbf,
                                                       const float* __restrict__ csqw,
                                                       long long* __restrict__ ids,
                                                       float* __restrict__ rec)
{
    __shared__ __bf16 s_res[TOK_PER_WG * DD];   // 128 KB residual (bf16)
    __shared__ __bf16 s_cb[2 * KK * 32];        // 2 x 16 KB codebook chunk
    __shared__ float  s_csq[KK];
    __shared__ int    s_idx[TOK_PER_WG];

    const int tid   = threadIdx.x;
    const int wave  = tid >> 5;
    const int lane  = tid & 31;
    const int laneM = lane & 15;
    const int hi    = lane >> 4;
    const int locTok0 = wave * 16;
    const long long wgTok = (long long)blockIdx.x * TOK_PER_WG;

    // ---- load feature tile -> bf16 residual in LDS ----
    for (int r = 0; r < 16; ++r) {
        const float4* src = (const float4*)feat + (wgTok + locTok0 + r) * (DD / 4);
        __bf16* dst = &s_res[(locTok0 + r) * DD];
        #pragma unroll
        for (int j = 0; j < 4; ++j) {
            int d4 = lane + 32 * j;
            float4 f = src[d4];
            v4bf o;
            o[0] = (__bf16)f.x; o[1] = (__bf16)f.y;
            o[2] = (__bf16)f.z; o[3] = (__bf16)f.w;
            *(v4bf*)(dst + d4 * 4) = o;
        }
    }

    #pragma unroll 1
    for (int layer = 0; layer < LL; ++layer) {
        v8f acc[16];
        const v8f vzero = {0.f, 0.f, 0.f, 0.f, 0.f, 0.f, 0.f, 0.f};
        #pragma unroll
        for (int t = 0; t < 16; ++t) acc[t] = vzero;

        float csq_acc = 0.f;

        __syncthreads();  // close previous layer (argmin reads of s_csq, buffer reads)
        if constexpr (PRECONV) {
            s_csq[tid] = csqw[layer * KK + tid];
            if (wave == 0)  // prologue DMA: chunk 0 -> buffer 0
                tdm_issue_chunk(cbbf, layer, 0, (unsigned)(uintptr_t)&s_cb[0]);
        }

        #pragma unroll 1
        for (int chunk = 0; chunk < 16; ++chunk) {
            int bufBase;
            if constexpr (PRECONV) {
                __syncthreads();  // waves done reading the buffer the next DMA overwrites
                if (wave == 0) {
                    if (chunk + 1 < 16) {
                        tdm_issue_chunk(cbbf, layer, chunk + 1,
                                        (unsigned)(uintptr_t)&s_cb[((chunk + 1) & 1) * KK * 32]);
                        __builtin_amdgcn_s_wait_tensorcnt(1);  // chunk's DMA done (in-order)
                    } else {
                        __builtin_amdgcn_s_wait_tensorcnt(0);
                    }
                }
                __syncthreads();  // staged chunk visible to all waves
                bufBase = (chunk & 1) * KK * 32;
            } else {
                __syncthreads();
                {   // self-converting staging: thread tid stages centroid tid
                    const float4* src =
                        (const float4*)(cb + ((size_t)layer * KK + tid) * DD + chunk * 32);
                    #pragma unroll
                    for (int j = 0; j < 8; ++j) {
                        float4 f = src[j];
                        csq_acc += f.x * f.x + f.y * f.y + f.z * f.z + f.w * f.w;
                        v4bf o;
                        o[0] = (__bf16)f.x; o[1] = (__bf16)f.y;
                        o[2] = (__bf16)f.z; o[3] = (__bf16)f.w;
                        *(v4bf*)&s_cb[tid * 32 + j * 4] = o;
                    }
                }
                __syncthreads();
                bufBase = 0;
            }

            // ---- A fragment: 16 tokens x 32 dims from LDS residual ----
            const __bf16* arow = &s_res[(locTok0 + laneM) * DD + chunk * 32];
            v8bf alo = *(const v8bf*)(arow + hi * 8);
            v8bf ahi = *(const v8bf*)(arow + 16 + hi * 8);
            v16bf afrag;
            #pragma unroll
            for (int i = 0; i < 8; ++i) { afrag[i] = alo[i]; afrag[8 + i] = ahi[i]; }

            // ---- 16 centroid tiles, B fragments prefetched TWO tiles ahead so
            // each WMMA consumes loads issued two WMMAs earlier (partial waits).
            auto loadB = [&](int t) -> v16bf {
                const __bf16* brow = &s_cb[bufBase + (t * 16 + laneM) * 32 + hi * 8];
                v8bf blo = *(const v8bf*)brow;
                v8bf bh2 = *(const v8bf*)(brow + 16);
                v16bf b;
                #pragma unroll
                for (int i = 0; i < 8; ++i) { b[i] = blo[i]; b[8 + i] = bh2[i]; }
                return b;
            };

            v16bf b0 = loadB(0);
            v16bf b1 = loadB(1);
            #pragma unroll
            for (int t = 0; t < 16; t += 2) {
                v16bf p0 = b0, p1 = b1;
                if (t + 2 < 16) p0 = loadB(t + 2);
                if (t + 3 < 16) p1 = loadB(t + 3);
                acc[t]     = __builtin_amdgcn_wmma_f32_16x16x32_bf16(
                    false, afrag, false, b0, (short)0, acc[t], false, false);
                acc[t + 1] = __builtin_amdgcn_wmma_f32_16x16x32_bf16(
                    false, afrag, false, b1, (short)0, acc[t + 1], false, false);
                b0 = p0; b1 = p1;
            }
        }

        if constexpr (!PRECONV) s_csq[tid] = csq_acc;
        __syncthreads();

        // ---- argmin over K: per-lane scan over tiles, then 16-lane butterfly ----
        float bestv[8];
        int   besti[8];
        #pragma unroll
        for (int t = 0; t < 16; ++t) {
            float cs = s_csq[t * 16 + laneM];
            int kidx = t * 16 + laneM;
            #pragma unroll
            for (int v = 0; v < 8; ++v) {
                float sc = cs - 2.0f * acc[t][v];
                if (t == 0 || sc < bestv[v]) { bestv[v] = sc; besti[v] = kidx; }
            }
        }
        #pragma unroll
        for (int m = 1; m < 16; m <<= 1) {
            #pragma unroll
            for (int v = 0; v < 8; ++v) {
                float ov = __shfl_xor(bestv[v], m, 32);
                int   oi = __shfl_xor(besti[v], m, 32);
                if (ov < bestv[v] || (ov == bestv[v] && oi < besti[v])) {
                    bestv[v] = ov; besti[v] = oi;
                }
            }
        }
        if (laneM == 0) {
            #pragma unroll
            for (int v = 0; v < 8; ++v) {
                int r = v + 8 * hi;
                s_idx[locTok0 + r] = besti[v];
                ids[(wgTok + locTok0 + r) * LL + layer] = (long long)besti[v];
            }
        }
        asm volatile("s_wait_dscnt 0x0" ::: "memory");  // idx visible to whole wave

        // ---- residual -= selected centroid ----
        for (int r = 0; r < 16; ++r) {
            int kk = s_idx[locTok0 + r];
            __bf16* rr = &s_res[(locTok0 + r) * DD];
            if constexpr (PRECONV) {
                const __bf16* cbr = cbbf + ((size_t)layer * KK + kk) * DD;
                #pragma unroll
                for (int j = 0; j < 2; ++j) {
                    int d8 = lane + 32 * j;   // 8-element groups
                    v8bf cv  = *(const v8bf*)(cbr + d8 * 8);
                    v8bf old = *(const v8bf*)(rr + d8 * 8);
                    v8bf nw;
                    #pragma unroll
                    for (int i = 0; i < 8; ++i)
                        nw[i] = (__bf16)((float)old[i] - (float)cv[i]);
                    *(v8bf*)(rr + d8 * 8) = nw;
                }
            } else {
                const float4* cbr = (const float4*)(cb + ((size_t)layer * KK + kk) * DD);
                #pragma unroll
                for (int j = 0; j < 4; ++j) {
                    int d4 = lane + 32 * j;
                    float4 cv = cbr[d4];
                    v4bf old = *(v4bf*)(rr + d4 * 4);
                    v4bf nw;
                    nw[0] = (__bf16)((float)old[0] - cv.x);
                    nw[1] = (__bf16)((float)old[1] - cv.y);
                    nw[2] = (__bf16)((float)old[2] - cv.z);
                    nw[3] = (__bf16)((float)old[3] - cv.w);
                    *(v4bf*)(rr + d4 * 4) = nw;
                }
            }
        }
    }

    // ---- reconstructed = feature - final residual ----
    for (int r = 0; r < 16; ++r) {
        const float4* fr = (const float4*)feat + (wgTok + locTok0 + r) * (DD / 4);
        float4* orow = (float4*)rec + (wgTok + locTok0 + r) * (DD / 4);
        const __bf16* rr = &s_res[(locTok0 + r) * DD];
        #pragma unroll
        for (int j = 0; j < 4; ++j) {
            int d4 = lane + 32 * j;
            float4 f = fr[d4];
            v4bf rs = *(const v4bf*)(rr + d4 * 4);
            float4 o;
            o.x = f.x - (float)rs[0];
            o.y = f.y - (float)rs[1];
            o.z = f.z - (float)rs[2];
            o.w = f.w - (float)rs[3];
            orow[d4] = o;
        }
    }
}

extern "C" void kernel_launch(void* const* d_in, const int* in_sizes, int n_in,
                              void* d_out, int out_size, void* d_ws, size_t ws_size,
                              hipStream_t stream) {
    (void)in_sizes; (void)n_in; (void)out_size;
    const float* feat = (const float*)d_in[0];  // [B, N, D] f32
    const float* cb   = (const float*)d_in[1];  // [L, K, D] f32
    // Output tuple (semantic_ids int64 [B,N,L], reconstructed f32 [B,N,D])
    // concatenated flat: int64 ids occupy 2 float-slots each at the front.
    long long* ids = (long long*)d_out;
    float*     rec = (float*)d_out + (size_t)BB * NN * LL * 2;

    dim3 grid((BB * NN) / TOK_PER_WG);  // 1024 workgroups
    const size_t cbbf_bytes = (size_t)LL * KK * DD * sizeof(__bf16);
    const size_t need = cbbf_bytes + (size_t)LL * KK * sizeof(float);

    if (ws_size >= need) {
        __bf16* cbbf = (__bf16*)d_ws;
        float*  csqw = (float*)((char*)d_ws + cbbf_bytes);
        rq_prep<<<dim3(LL * KK), dim3(32), 0, stream>>>(cb, cbbf, csqw);
        rq_fused<true><<<grid, THREADS, 0, stream>>>(feat, cb, cbbf, csqw, ids, rec);
    } else {
        rq_fused<false><<<grid, THREADS, 0, stream>>>(feat, cb, nullptr, nullptr, ids, rec);
    }
}
